// S4Predictor_56367150792785
// MI455X (gfx1250) — matrix operand order repaired and checked by reading, not compiled, not run on hardware
//
#include <hip/hip_runtime.h>
#include <hip/hip_bf16.h>

typedef __attribute__((ext_vector_type(16))) _Float16 v16h;
typedef __attribute__((ext_vector_type(8)))  _Float16 v8h;
typedef __attribute__((ext_vector_type(8)))  float    v8f;
typedef __attribute__((ext_vector_type(4)))  unsigned int u32x4;
typedef __attribute__((ext_vector_type(8)))  int i32x8;
typedef __attribute__((ext_vector_type(4)))  int i32x4;

#define B_SZ   2048
#define L_SEQ  168
#define H_DIM  96
#define NBLK   4
#define N2     48
#define BL     (B_SZ * L_SEQ)          // 344064
#define NE     ((size_t)BL * H_DIM)    // 33030144
#define LPAD   176                     // 11 tiles of 16
#define KPAD   192                     // 6 chunks of 32

#if __has_builtin(__builtin_amdgcn_tensor_load_to_lds) && \
    __has_builtin(__builtin_amdgcn_s_wait_tensorcnt)
#define USE_TDM_BUILTIN 1
#endif

// ---------------------------------------------------------------------------
// WMMA fragment index helpers (contiguous k order used by this file, not the ISA table order; wave32)
// A (16x32 f16): lane -> row M = lane&15; element j -> K = (lane>=16?16:0)+j
__device__ __forceinline__ int a_kidx(int j, int lh) {
    return lh * 16 + j;
}
// B (32x16 f16): lane -> col N = lane&15; element j -> K = (lane>=16?16:0)+j
// C/D (16x16 f32, v8f): element r -> M = (lane>=16?8:0)+r ; N = lane&15

__device__ __forceinline__ float fast_tanh(float x) {
#if __has_builtin(__builtin_amdgcn_tanhf)
    return __builtin_amdgcn_tanhf(x);
#else
    const float xc = fminf(fmaxf(x, -10.0f), 10.0f);
    const float e  = __expf(2.0f * xc);
    return (e - 1.0f) / (e + 1.0f);
#endif
}

__device__ __forceinline__ unsigned int lds_off32(const void* p) {
    return (unsigned int)(uintptr_t)p;   // flat LDS addr: offset in low 32 bits
}

#ifdef USE_TDM_BUILTIN
// 2-byte-element 2D tile load Global->LDS via Tensor Data Mover.
// D# per CDNA5 ISA: group0 {count=1, lds_addr,
// global_addr, type=2}; group1 {data_size=1(2B), tensor_dim0/1, tile_dim0/1,
// tensor_dim0_stride}. Groups 2/3 zero (<=2D tensor).
__device__ __forceinline__ void tdm_load_2d(unsigned int lds_addr,
                                            const void* gptr,
                                            unsigned int tile0,
                                            unsigned int tile1,
                                            unsigned long long dim0,
                                            unsigned long long dim1,
                                            unsigned long long stride0) {
    const unsigned long long ga = (unsigned long long)(uintptr_t)gptr;
    u32x4 g0 = {};
    g0[0] = 1u;                                        // count=1 (valid D#)
    g0[1] = lds_addr;                                  // lds_addr (bytes)
    g0[2] = (unsigned int)ga;                          // global_addr[31:0]
    g0[3] = (unsigned int)((ga >> 32) & 0x01ffffffu)   // global_addr[56:32]
          | (2u << 30);                                // type=2 ("image")
    i32x8 g1 = {};
    g1[0] = (int)(1u << 16);                           // data_size=1 -> 2 bytes
    g1[1] = (int)(((unsigned int)(dim0 & 0xffffu)) << 16);          // dim0[15:0]
    g1[2] = (int)(((unsigned int)((dim0 >> 16) & 0xffffu))          // dim0[31:16]
          | (((unsigned int)(dim1 & 0xffffu)) << 16));              // dim1[15:0]
    g1[3] = (int)(((unsigned int)((dim1 >> 16) & 0xffffu))          // dim1[31:16]
          | (tile0 << 16));                                         // tile_dim0
    g1[4] = (int)tile1;                                // tile_dim1 (tile_dim2=0)
    g1[5] = (int)(unsigned int)(stride0 & 0xffffffffu);// dim0_stride[31:0]
    g1[6] = (int)(unsigned int)((stride0 >> 32) & 0xffffu); // dim0_stride[47:32]
    g1[7] = 0;
    i32x4 gz = {};
#if defined(__clang_major__) && (__clang_major__ >= 23)
    i32x8 gz8 = {};
    __builtin_amdgcn_tensor_load_to_lds(g0, g1, gz, gz, gz8, 0);
#else
    __builtin_amdgcn_tensor_load_to_lds(g0, g1, gz, gz, 0);
#endif
}
#endif

// ---------------------------------------------------------------------------
// 1) S4D kernel K[i][h][l]  (ZOH discretization, complex pairs, 2*Re)
__global__ void s4d_K_kernel(const float* __restrict__ log_dt,
                             const float* __restrict__ log_A_real,
                             const float* __restrict__ A_imag,
                             const float* __restrict__ C_re,
                             const float* __restrict__ C_im,
                             float* __restrict__ Kf) {
    const int blk = blockIdx.y;          // 0..3
    const int hc  = blockIdx.x;          // 0..95
    const int l   = threadIdx.x;         // 0..167
    const float dt = __expf(log_dt[blk * H_DIM + hc]);
    float acc = 0.0f;
    const int base0 = (blk * H_DIM + hc) * N2;
    for (int n = 0; n < N2; ++n) {
        const int base = base0 + n;
        const float Ar = -__expf(log_A_real[base]);
        const float Ai = A_imag[base];
        const float dr = dt * Ar, di = dt * Ai;
        const float er = __expf(dr);
        const float eAr = er * __cosf(di), eAi = er * __sinf(di);
        const float nr = eAr - 1.0f, ni = eAi;
        const float d2 = 1.0f / (Ar * Ar + Ai * Ai);
        const float qr = (nr * Ar + ni * Ai) * d2;
        const float qi = (ni * Ar - nr * Ai) * d2;
        const float Cr = C_re[base], Ci = C_im[base];
        const float Ctr = Cr * qr - Ci * qi;
        const float Cti = Cr * qi + Ci * qr;
        const float fl = (float)l;
        const float vm = __expf(dr * fl);
        const float Vr = vm * __cosf(di * fl);
        const float Vi = vm * __sinf(di * fl);
        acc += Ctr * Vr - Cti * Vi;
    }
    Kf[(blk * H_DIM + hc) * L_SEQ + l] = 2.0f * acc;
}

// ---------------------------------------------------------------------------
// 2) Toeplitz, stored TRANSPOSED for contiguous B-fragments:
//    T2[ch][l][t] = K[ch][l-t]  (t contiguous, KPAD stride)
__global__ void build_T_kernel(const float* __restrict__ Kf,
                               _Float16* __restrict__ T2, int blk) {
    const int total = H_DIM * LPAD * KPAD;
    int idx = blockIdx.x * blockDim.x + threadIdx.x;
    if (idx >= total) return;
    const int t  = idx % KPAD;
    const int l  = (idx / KPAD) % LPAD;
    const int ch = idx / (KPAD * LPAD);
    float v = 0.0f;
    if (t <= l && l < L_SEQ && t < L_SEQ)
        v = Kf[(blk * H_DIM + ch) * L_SEQ + (l - t)];
    T2[idx] = (_Float16)v;
}

// ---------------------------------------------------------------------------
// 3) Encoder: channel-major f32 hT + f16 shadow hTh
__global__ void encoder_kernel(const float* __restrict__ x,
                               const float* __restrict__ W,
                               const float* __restrict__ b,
                               float* __restrict__ hT,
                               _Float16* __restrict__ hTh) {
    size_t idx = (size_t)blockIdx.x * blockDim.x + threadIdx.x;
    if (idx >= NE) return;
    const int m  = (int)(idx % BL);
    const int ch = (int)(idx / BL);
    float s = b[ch];
#pragma unroll
    for (int d = 0; d < 9; ++d)
        s += x[(size_t)m * 9 + d] * W[d * H_DIM + ch];
    hT[idx]  = s;
    hTh[idx] = (_Float16)s;
}

// ---------------------------------------------------------------------------
// 4) Causal conv (per-channel Toeplitz WMMA) + fused skip(D*u) + GELU -> f16 zT
//    A fragments: aligned v8h loads, fully register-resident (unrolled).
//    B panel (16 l-rows x 192 t) staged to LDS once per tile via TDM and
//    shared by all 4 waves; ds_load_b128 fragments.
__global__ __launch_bounds__(128) void s4_conv_kernel(
    const float* __restrict__ hT, const _Float16* __restrict__ hTh,
    const _Float16* __restrict__ T2, const float* __restrict__ Dskip,
    _Float16* __restrict__ zT, int blk) {
    __shared__ __align__(32) _Float16 ldsB[16 * KPAD];   // 6 KB

    const int ch    = blockIdx.x;                 // 0..95
    const int wid   = threadIdx.x >> 5;
    const int lane  = threadIdx.x & 31;
    const int lh    = lane >> 4, l16 = lane & 15;
    const int btile = blockIdx.y * 4 + wid;       // 0..127
    const int b0    = btile * 16;

    const _Float16* rowA = hTh + ((size_t)ch * B_SZ + (b0 + l16)) * L_SEQ;
    v16h afrag[6];
#pragma unroll
    for (int kc = 0; kc < 5; ++kc) {              // t < 160 < L_SEQ: unguarded
        const v8h lo = *(const v8h*)(rowA + kc * 32 + lh * 16);
        const v8h hi = *(const v8h*)(rowA + kc * 32 + lh * 16 + 8);
#pragma unroll
        for (int j = 0; j < 8; ++j) { afrag[kc][j] = lo[j]; afrag[kc][j + 8] = hi[j]; }
    }
    // tail chunk: only lanes 0-15, j<8 have t=160..167 in range
#pragma unroll
    for (int j = 0; j < 16; ++j) afrag[5][j] = (_Float16)0.0f;
    if (lh == 0) {
        const v8h lo = *(const v8h*)(rowA + 160);
#pragma unroll
        for (int j = 0; j < 8; ++j) afrag[5][j] = lo[j];
    }

    const float dsk = Dskip[blk * H_DIM + ch];
#pragma unroll
    for (int nt = 0; nt < 11; ++nt) {
        const int lmax = nt * 16 + 15;
        const int tmax = (lmax < L_SEQ - 1) ? lmax : (L_SEQ - 1);
        const int kcn  = (tmax >> 5) + 1;         // causality bound (folds)
#ifdef USE_TDM_BUILTIN
        __syncthreads();                          // LDS reuse from prev tile
        if (threadIdx.x < 32) {                   // one wave issues the DMA
            tdm_load_2d(lds_off32(ldsB),
                        T2 + ((size_t)ch * LPAD + nt * 16) * KPAD,
                        /*tile0=*/16 * KPAD, /*tile1=*/0,
                        /*dim0=*/(unsigned long long)LPAD * KPAD, /*dim1=*/1,
                        /*stride0=*/16 * KPAD);
            __builtin_amdgcn_s_wait_tensorcnt(0);
        }
        __syncthreads();
#else
        const _Float16* rowB = T2 + ((size_t)ch * LPAD + nt * 16 + l16) * KPAD;
#endif
        v8f acc = {};
#pragma unroll
        for (int kc = 0; kc < 6; ++kc) {
            if (kc < kcn) {                       // uniform, constant-folds
#ifdef USE_TDM_BUILTIN
                const v16h bfrag =
                    *(const v16h*)(&ldsB[l16 * KPAD + kc * 32 + lh * 16]);
#else
                const v16h bfrag = *(const v16h*)(rowB + kc * 32 + lh * 16);
#endif
                acc = __builtin_amdgcn_wmma_f32_16x16x32_f16(
                    false, afrag[kc], false, bfrag, (short)0, acc, false, false);
            }
        }
        const int l = nt * 16 + l16;
        if (l < L_SEQ) {
#pragma unroll
            for (int r = 0; r < 8; ++r) {
                const int bb = b0 + lh * 8 + r;
                const size_t o = ((size_t)ch * B_SZ + bb) * L_SEQ + l;
                float v = acc[r] + hT[o] * dsk;       // conv + D*u skip
                const float u = v * (1.0f + 0.044715f * v * v);
                const float g = 0.5f * v *
                    (1.0f + fast_tanh(0.7978845608028654f * u));
                zT[o] = (_Float16)g;
            }
        }
    }
}

// ---------------------------------------------------------------------------
// 5) N=96 GEMM, 64-row M-tiles (4 sub-tiles/wave), A-chunk staged via TDM when
//    f16 (VALU fallback / f32 path), W chunk in LDS; optional fused
//    bias+residual+LayerNorm updating f32 hT and its f16 shadow.
//    LDS A layout: ATRANS -> [k][64] (TDM tile rows), else [m][32].
template <typename AT, typename CT, bool ATRANS, bool FUSE_LN>
__global__ __launch_bounds__(192) void gemm96_kernel(
    const AT* __restrict__ A, const float* __restrict__ W,
    const float* __restrict__ bias, const float* __restrict__ lnw,
    const float* __restrict__ lnb, float* __restrict__ hres,
    _Float16* __restrict__ hres16, CT* __restrict__ Cout,
    int K, int Mstride) {
    __shared__ __align__(32) _Float16 lsa[64 * 32];
    __shared__ _Float16 lsw[32][100];
    __shared__ float lout[FUSE_LN ? 64 : 1][100];
    __shared__ float smu[FUSE_LN ? 64 : 1], srs[FUSE_LN ? 64 : 1];

#ifdef USE_TDM_BUILTIN
    constexpr bool kTdm = (sizeof(AT) == 2);
#else
    constexpr bool kTdm = false;
#endif

    const int m0   = blockIdx.x * 64;
    const int tid  = threadIdx.x;
    const int wid  = tid >> 5;
    const int lane = tid & 31;
    const int lh   = lane >> 4, l16 = lane & 15;
    const int n0   = wid * 16;

    v8f acc[4] = {};
    const int nkc = K >> 5;
    for (int kc = 0; kc < nkc; ++kc) {
        __syncthreads();
        if constexpr (kTdm) {
#ifdef USE_TDM_BUILTIN
            if (tid < 32) {
                if (ATRANS)   // 64 contiguous m x 32 k-rows -> LDS [k][64]
                    tdm_load_2d(lds_off32(lsa),
                                A + (size_t)(kc * 32) * Mstride + m0,
                                /*tile0=*/64, /*tile1=*/32,
                                /*dim0=*/(unsigned long long)Mstride,
                                /*dim1=*/(unsigned long long)K,
                                /*stride0=*/(unsigned long long)Mstride);
                else          // 32 contiguous k x 64 m-rows -> LDS [m][32]
                    tdm_load_2d(lds_off32(lsa),
                                A + (size_t)m0 * K + kc * 32,
                                /*tile0=*/32, /*tile1=*/64,
                                /*dim0=*/(unsigned long long)K,
                                /*dim1=*/(unsigned long long)Mstride,
                                /*stride0=*/(unsigned long long)K);
                __builtin_amdgcn_s_wait_tensorcnt(0);
            }
#endif
        } else {
            for (int idx = tid; idx < 64 * 32; idx += 192) {
                int m, k;
                if (ATRANS) { m = idx & 63; k = idx >> 6; }
                else        { k = idx & 31; m = idx >> 5; }
                const _Float16 v = (_Float16)(ATRANS
                    ? A[(size_t)(kc * 32 + k) * Mstride + m0 + m]
                    : A[(size_t)(m0 + m) * K + kc * 32 + k]);
                lsa[ATRANS ? (k * 64 + m) : (m * 32 + k)] = v;
            }
        }
        for (int idx = tid; idx < 32 * H_DIM; idx += 192) {
            const int k = idx / H_DIM, n = idx - k * H_DIM;
            lsw[k][n] = (_Float16)W[(size_t)(kc * 32 + k) * H_DIM + n];
        }
        __syncthreads();
        v16h bf;
#pragma unroll
        for (int j = 0; j < 16; ++j) bf[j] = lsw[lh * 16 + j][n0 + l16];
#pragma unroll
        for (int s = 0; s < 4; ++s) {
            v16h af;
#pragma unroll
            for (int j = 0; j < 16; ++j)
                af[j] = ATRANS ? lsa[a_kidx(j, lh) * 64 + s * 16 + l16]
                               : lsa[(s * 16 + l16) * 32 + a_kidx(j, lh)];
            acc[s] = __builtin_amdgcn_wmma_f32_16x16x32_f16(
                false, af, false, bf, (short)0, acc[s], false, false);
        }
    }

    const float bv = bias[n0 + l16];
    if (!FUSE_LN) {
#pragma unroll
        for (int s = 0; s < 4; ++s)
#pragma unroll
            for (int r = 0; r < 8; ++r)
                Cout[(size_t)(m0 + s * 16 + lh * 8 + r) * H_DIM + n0 + l16] =
                    (CT)(acc[s][r] + bv);
    } else {
#pragma unroll
        for (int s = 0; s < 4; ++s)
#pragma unroll
            for (int r = 0; r < 8; ++r)
                lout[s * 16 + lh * 8 + r][n0 + l16] = acc[s][r] + bv;
        __syncthreads();
        for (int idx = tid; idx < 64 * H_DIM; idx += 192) {
            const int m = idx & 63, n = idx >> 6;
            lout[m][n] += hres[(size_t)n * Mstride + m0 + m];
        }
        __syncthreads();
        if (tid < 64) {
            float s = 0.0f, ss = 0.0f;
            for (int n = 0; n < H_DIM; ++n) {
                const float v = lout[tid][n];
                s += v; ss += v * v;
            }
            const float mu  = s * (1.0f / H_DIM);
            const float var = ss * (1.0f / H_DIM) - mu * mu;
            smu[tid] = mu;
            srs[tid] = rsqrtf(var + 1e-5f);
        }
        __syncthreads();
        for (int idx = tid; idx < 64 * H_DIM; idx += 192) {
            const int m = idx & 63, n = idx >> 6;
            const float o = (lout[m][n] - smu[m]) * srs[m] * lnw[n] + lnb[n];
            hres  [(size_t)n * Mstride + m0 + m] = o;
            hres16[(size_t)n * Mstride + m0 + m] = (_Float16)o;
        }
    }
}

// ---------------------------------------------------------------------------
extern "C" void kernel_launch(void* const* d_in, const int* in_sizes, int n_in,
                              void* d_out, int out_size, void* d_ws, size_t ws_size,
                              hipStream_t stream) {
    const float* x          = (const float*)d_in[0];
    const float* enc_W      = (const float*)d_in[1];
    const float* enc_b      = (const float*)d_in[2];
    const float* log_dt     = (const float*)d_in[3];
    const float* log_A_real = (const float*)d_in[4];
    const float* A_imag     = (const float*)d_in[5];
    const float* C_re       = (const float*)d_in[6];
    const float* C_im       = (const float*)d_in[7];
    const float* D_skip     = (const float*)d_in[8];
    const float* W_out      = (const float*)d_in[9];
    const float* b_out      = (const float*)d_in[10];
    const float* ln_w       = (const float*)d_in[11];
    const float* ln_b       = (const float*)d_in[12];
    const float* dec_W      = (const float*)d_in[13];
    const float* dec_b      = (const float*)d_in[14];
    const float* fc1_W      = (const float*)d_in[15];
    const float* fc1_b      = (const float*)d_in[16];
    const float* fc2_W      = (const float*)d_in[17];
    const float* fc2_b      = (const float*)d_in[18];

    char* ws = (char*)d_ws;
    float*    hT   = (float*)(ws);                              // f32 [H][B][L]
    _Float16* hTh  = (_Float16*)(ws + 132120576);               // f16 shadow
    _Float16* zT   = (_Float16*)(ws + 198180864);               // f16 z / dec out
    float*    Kf   = (float*)(ws + 264241152);                  // 4*96*168
    _Float16* T2   = (_Float16*)(ws + 264499200);               // 96*176*192 f16
    float*    fc1o = (float*)(ws + 270987264);                  // 2048*96

    s4d_K_kernel<<<dim3(H_DIM, NBLK), L_SEQ, 0, stream>>>(
        log_dt, log_A_real, A_imag, C_re, C_im, Kf);

    encoder_kernel<<<(unsigned)((NE + 255) / 256), 256, 0, stream>>>(
        x, enc_W, enc_b, hT, hTh);

    const int NT = H_DIM * LPAD * KPAD;
    for (int i = 0; i < NBLK; ++i) {
        build_T_kernel<<<(NT + 255) / 256, 256, 0, stream>>>(Kf, T2, i);
        s4_conv_kernel<<<dim3(H_DIM, B_SZ / 64), 128, 0, stream>>>(
            hT, hTh, T2, D_skip, zT, i);
        gemm96_kernel<_Float16, float, true, true><<<BL / 64, 192, 0, stream>>>(
            zT, W_out + i * H_DIM * H_DIM, b_out + i * H_DIM,
            ln_w + i * H_DIM, ln_b + i * H_DIM, hT, hTh, (float*)nullptr,
            H_DIM, BL);
    }

    // Decoder: hT (transposed f32) @ dec_W -> f16 row-major (BL x 96) in zT
    gemm96_kernel<float, _Float16, true, false><<<BL / 64, 192, 0, stream>>>(
        hT, dec_W, dec_b, nullptr, nullptr, nullptr, nullptr, zT, H_DIM, BL);

    // fc1: f16 (2048 x 16128) @ fc1_W -> f32 fc1o
    gemm96_kernel<_Float16, float, false, false><<<B_SZ / 64, 192, 0, stream>>>(
        zT, fc1_W, fc1_b, nullptr, nullptr, nullptr, nullptr, fc1o,
        L_SEQ * H_DIM, B_SZ);

    // fc2: (2048 x 96) @ fc2_W -> d_out
    gemm96_kernel<float, float, false, false><<<B_SZ / 64, 192, 0, stream>>>(
        fc1o, fc2_W, fc2_b, nullptr, nullptr, nullptr, nullptr, (float*)d_out,
        H_DIM, B_SZ);

    (void)in_sizes; (void)n_in; (void)out_size; (void)ws_size;
}
